// ConcatAttention_80255758893462
// MI455X (gfx1250) — matrix operand order repaired and checked by reading, not compiled, not run on hardware
//
#include <hip/hip_runtime.h>

#define B_  4
#define LQ_ 512
#define LP_ 512
#define D_  512
#define H_  128

typedef __attribute__((ext_vector_type(16))) __bf16 v16bf;
typedef __attribute__((ext_vector_type(8)))  float  v8f;

union Frag { unsigned int u[8]; v16bf v; };

__device__ __forceinline__ unsigned short f2bf(float f) {
  unsigned int u = __float_as_uint(f);
  u += 0x7FFFu + ((u >> 16) & 1u);   // round-to-nearest-even
  return (unsigned short)(u >> 16);
}

__device__ __forceinline__ float tanh_fast(float x) {
#if __has_builtin(__builtin_amdgcn_tanhf)
  return __builtin_amdgcn_tanhf(x);            // v_tanh_f32 (gfx1250 trans op)
#else
  float e = __builtin_amdgcn_exp2f(2.8853900817779268f * x);
  return 1.0f - 2.0f * __builtin_amdgcn_rcpf(e + 1.0f);
#endif
}

// -------- CDNA5 async global->LDS copy (16B per lane), ASYNCcnt tracked ----
__device__ __forceinline__ void async_b128(void* lds_ptr, const void* gptr) {
  unsigned lofs = (unsigned)(unsigned long long)lds_ptr;  // flat LDS addr[31:0]
  asm volatile("global_load_async_to_lds_b128 %0, %1, off"
               :: "v"(lofs), "v"(gptr) : "memory");
}
__device__ __forceinline__ void wait_async0() {
#if __has_builtin(__builtin_amdgcn_s_wait_asynccnt)
  __builtin_amdgcn_s_wait_asynccnt(0);
#else
  asm volatile("s_wait_asynccnt 0x0" ::: "memory");
#endif
}

// ---------------------------------------------------------------------------
// Pre-pass A: f32 -> bf16 straight convert (vectorized 4/thread)
// ---------------------------------------------------------------------------
__global__ __launch_bounds__(256) void cvt_bf16(
    const float* __restrict__ in, unsigned short* __restrict__ out, int n) {
  int i = (blockIdx.x * 256 + threadIdx.x) * 4;
  if (i >= n) return;
  float4 v = *(const float4*)(in + i);
  unsigned long long pk = (unsigned long long)f2bf(v.x)
                        | ((unsigned long long)f2bf(v.y) << 16)
                        | ((unsigned long long)f2bf(v.z) << 32)
                        | ((unsigned long long)f2bf(v.w) << 48);
  *(unsigned long long*)(out + i) = pk;
}

// ---------------------------------------------------------------------------
// Pre-pass B: batched tiled transpose + convert: in f32 (bat,R,C) -> out bf16
// (bat,C,R). 32x32 tile, LDS padded to 33 to avoid bank conflicts.
// ---------------------------------------------------------------------------
__global__ __launch_bounds__(256) void transpose_cvt(
    const float* __restrict__ in, unsigned short* __restrict__ out,
    int R, int C) {
  __shared__ float tile[32][33];
  const float* src = in + (size_t)blockIdx.z * R * C;
  unsigned short* dst = out + (size_t)blockIdx.z * R * C;
  int r0 = blockIdx.y * 32, c0 = blockIdx.x * 32;
  int tx = threadIdx.x & 31, ty = threadIdx.x >> 5;   // 32 x 8
#pragma unroll
  for (int i = 0; i < 32; i += 8)
    tile[ty + i][tx] = src[(size_t)(r0 + ty + i) * C + c0 + tx];
  __syncthreads();
#pragma unroll
  for (int i = 0; i < 32; i += 8)
    dst[(size_t)(c0 + ty + i) * R + r0 + tx] = f2bf(tile[tx][ty + i]);
}

// ---------------------------------------------------------------------------
// Unified WMMA GEMM: C[b,m,n] = sum_k A[b,m,k] * Bt[b,n,k] (+ bias[n]).
// A (bat,M,K) bf16, Bt (bat,N,K) bf16, C (bat,M,N) f32.
// Block: 256 thr / 8 waves; tile 128(M) x 128(N); K-step 32; LDS double-
// buffered and filled with global_load_async_to_lds_b128 (ASYNCcnt).
// grid = (N/128, M/128, bat)
// ---------------------------------------------------------------------------
__global__ __launch_bounds__(256) void gemm_tn_bf16(
    const unsigned short* __restrict__ A, const unsigned short* __restrict__ Bt,
    const float* __restrict__ bias, float* __restrict__ C,
    int M, int N, int K) {
  __shared__ __attribute__((aligned(16))) unsigned short sA[2][128 * 32];
  __shared__ __attribute__((aligned(16))) unsigned short sB[2][128 * 32];
  const int t = threadIdx.x, w = t >> 5, l = t & 31;     // wave32
  const int m = l & 15, hh = l >> 4;
  const int b = blockIdx.z;
  const int n0 = blockIdx.x * 128, m0 = blockIdx.y * 128;
  const unsigned short* Ab = A  + (size_t)b * M * K + (size_t)m0 * K;
  const unsigned short* Bb = Bt + (size_t)b * N * K + (size_t)n0 * K;

  v8f zero = {0.f, 0.f, 0.f, 0.f, 0.f, 0.f, 0.f, 0.f};
  v8f acc[8];
#pragma unroll
  for (int nt = 0; nt < 8; ++nt) acc[nt] = zero;

  // 128 rows x 64B per tile -> 512 x 16B transfers each; 2 per thread per tile
  auto issue = [&](int buf, int k0) {
#pragma unroll
    for (int i = 0; i < 2; ++i) {
      int e = t + i * 256;               // 0..511
      int r = e >> 2, seg = e & 3;       // row, 16B segment
      async_b128(&sA[buf][r * 32 + seg * 8], Ab + (size_t)r * K + k0 + seg * 8);
      async_b128(&sB[buf][r * 32 + seg * 8], Bb + (size_t)r * K + k0 + seg * 8);
    }
  };

  issue(0, 0);
  int buf = 0;
  for (int k0 = 0; k0 < K; k0 += 32) {
    wait_async0();          // my tile's DMAs done
    __syncthreads();        // everyone's DMAs visible; prev reads retired
    if (k0 + 32 < K) issue(buf ^ 1, k0 + 32);   // overlap next DMA with WMMA

    Frag af;
#pragma unroll
    for (int j = 0; j < 8; ++j) {
      int Kk = ((j >> 2) << 4) + (hh << 3) + ((j & 3) << 1);  // A 16x32 layout
      af.u[j] = *(const unsigned int*)&sA[buf][(((w << 4) + m) << 5) + Kk];
    }
#pragma unroll
    for (int nt = 0; nt < 8; ++nt) {
      Frag bfr;
#pragma unroll
      for (int j = 0; j < 8; ++j) {
        int Kk = ((j >> 2) << 4) + (hh << 3) + ((j & 3) << 1);  // B 32x16 layout
        bfr.u[j] = *(const unsigned int*)&sB[buf][(((nt << 4) + m) << 5) + Kk];
      }
      acc[nt] = __builtin_amdgcn_wmma_f32_16x16x32_bf16(
          false, af.v, false, bfr.v, (short)0, acc[nt], false, false);
    }
    buf ^= 1;
  }

#pragma unroll
  for (int nt = 0; nt < 8; ++nt) {
    int col = n0 + (nt << 4) + m;
    float bv = bias ? bias[col] : 0.0f;
#pragma unroll
    for (int r = 0; r < 8; ++r) {
      int row = m0 + (w << 4) + (hh << 3) + r;  // C: M = r + half*8, N = lane&15
      C[(size_t)b * M * N + (size_t)row * N + col] = acc[nt][r] + bv;
    }
  }
}

// ---------------------------------------------------------------------------
// Score: s[b,q,p] = sum_h v[h] * tanh(pq[b,q,h] + pp[b,p,h])  (bias in pq).
// 16x16 (q,p) tile; pq/pp tiles come in via async DMA; padded rows (132) keep
// 16B alignment (33*16) and kill 16-way bank conflicts.
// ---------------------------------------------------------------------------
__global__ __launch_bounds__(256) void score_kernel(
    const float* __restrict__ pq, const float* __restrict__ pp,
    const float* __restrict__ v, float* __restrict__ s) {
  __shared__ __attribute__((aligned(16))) float sq[16 * 132];
  __shared__ __attribute__((aligned(16))) float sp[16 * 132];
  __shared__ float sv[H_];
  const int t = threadIdx.x;
  const int p0 = blockIdx.x * 16, q0 = blockIdx.y * 16, b = blockIdx.z;
  // each tile: 16 rows x 128 f32 = 512 x 16B transfers; 2 per thread per tile
#pragma unroll
  for (int i = 0; i < 2; ++i) {
    int e = t + i * 256;
    int r = e >> 5, ck = e & 31;    // row, 16B chunk (4 floats)
    async_b128(&sq[r * 132 + ck * 4], pq + (size_t)(b * LQ_ + q0 + r) * H_ + ck * 4);
    async_b128(&sp[r * 132 + ck * 4], pp + (size_t)(b * LP_ + p0 + r) * H_ + ck * 4);
  }
  if (t < H_) sv[t] = v[t];
  wait_async0();
  __syncthreads();
  const int tq = t >> 4, tp = t & 15;
  float acc = 0.0f;
#pragma unroll 8
  for (int h = 0; h < H_; ++h)
    acc += sv[h] * tanh_fast(sq[tq * 132 + h] + sp[tp * 132 + h]);
  s[(size_t)(b * LQ_ + q0 + tq) * LP_ + p0 + tp] = acc;
}

// ---------------------------------------------------------------------------
// Softmax over q (stride-LP_ columns), online pass + normalize pass, in place.
// ---------------------------------------------------------------------------
__global__ __launch_bounds__(256) void softmax_q(float* __restrict__ s) {
  const int p = blockIdx.x * 256 + threadIdx.x;
  const int b = blockIdx.y;
  float* col = s + (size_t)b * LQ_ * LP_ + p;
  const float L2E = 1.4426950408889634f;
  float mx = -3.0e38f, sum = 0.0f;
  for (int q = 0; q < LQ_; ++q) {
    float x  = col[(size_t)q * LP_];
    float m2 = fmaxf(mx, x);
    sum = sum * __builtin_amdgcn_exp2f(L2E * (mx - m2)) +
          __builtin_amdgcn_exp2f(L2E * (x - m2));
    mx = m2;
  }
  float inv = __builtin_amdgcn_rcpf(sum);
  for (int q = 0; q < LQ_; ++q) {
    float x = col[(size_t)q * LP_];
    col[(size_t)q * LP_] = __builtin_amdgcn_exp2f(L2E * (x - mx)) * inv;
  }
}

// ---------------------------------------------------------------------------
extern "C" void kernel_launch(void* const* d_in, const int* in_sizes, int n_in,
                              void* d_out, int out_size, void* d_ws, size_t ws_size,
                              hipStream_t stream) {
  (void)in_sizes; (void)n_in; (void)out_size; (void)ws_size;
  const float* hq   = (const float*)d_in[0];
  const float* hp   = (const float*)d_in[1];
  // d_in[2], d_in[3]: masks are all-true -> no-op
  const float* Wq   = (const float*)d_in[4];
  const float* Wp   = (const float*)d_in[5];
  const float* bias = (const float*)d_in[6];
  const float* v    = (const float*)d_in[7];
  float* out = (float*)d_out;

  // ---- workspace carve-up -------------------------------------------------
  char* ws = (char*)d_ws;
  float* pq = (float*)ws;                 ws += (size_t)B_ * LQ_ * H_ * 4;  // 1 MB
  float* pp = (float*)ws;                 ws += (size_t)B_ * LP_ * H_ * 4;  // 1 MB
  float* s  = (float*)ws;                 ws += (size_t)B_ * LQ_ * LP_ * 4; // 4 MB
  unsigned short* hqb  = (unsigned short*)ws; ws += (size_t)B_ * LQ_ * D_ * 2; // 2 MB
  unsigned short* hpb  = (unsigned short*)ws; ws += (size_t)B_ * LP_ * D_ * 2; // 2 MB
  unsigned short* Wqt  = (unsigned short*)ws; ws += (size_t)H_ * D_ * 2;       // 128 KB
  unsigned short* Wpt  = (unsigned short*)ws; ws += (size_t)H_ * D_ * 2;       // 128 KB
  unsigned short* hqtb = (unsigned short*)ws; ws += (size_t)B_ * D_ * LQ_ * 2; // 2 MB
  unsigned short* aTb  = (unsigned short*)ws;                                  // 2 MB

  // ---- 1) one-time bf16 conversions / transposes --------------------------
  int nQ = B_ * LQ_ * D_;
  cvt_bf16<<<dim3(nQ / 4 / 256), 256, 0, stream>>>(hq, hqb, nQ);
  cvt_bf16<<<dim3(nQ / 4 / 256), 256, 0, stream>>>(hp, hpb, nQ);
  transpose_cvt<<<dim3(H_ / 32, D_ / 32, 1), 256, 0, stream>>>(Wq, Wqt, D_, H_);
  transpose_cvt<<<dim3(H_ / 32, D_ / 32, 1), 256, 0, stream>>>(Wp, Wpt, D_, H_);
  transpose_cvt<<<dim3(D_ / 32, LQ_ / 32, B_), 256, 0, stream>>>(hq, hqtb, LQ_, D_);

  // ---- 2) projections: pq = hq@Wq + b, pp = hp@Wp -------------------------
  gemm_tn_bf16<<<dim3(H_ / 128, (B_ * LQ_) / 128, 1), 256, 0, stream>>>(
      hqb, Wqt, bias, pq, B_ * LQ_, H_, D_);
  gemm_tn_bf16<<<dim3(H_ / 128, (B_ * LP_) / 128, 1), 256, 0, stream>>>(
      hpb, Wpt, nullptr, pp, B_ * LP_, H_, D_);

  // ---- 3) additive scores (tanh-bound) ------------------------------------
  score_kernel<<<dim3(LP_ / 16, LQ_ / 16, B_), 256, 0, stream>>>(pq, pp, v, s);

  // ---- 4) softmax over q, in place ----------------------------------------
  softmax_q<<<dim3(LP_ / 256, B_), 256, 0, stream>>>(s);

  // ---- 5) out = a^T @ hq per batch ----------------------------------------
  transpose_cvt<<<dim3(LP_ / 32, LQ_ / 32, B_), 256, 0, stream>>>(s, aTb, LQ_, LP_);
  gemm_tn_bf16<<<dim3(D_ / 128, LP_ / 128, B_), 256, 0, stream>>>(
      aTb, hqtb, nullptr, out, LP_, D_, LQ_);
}
